// GraphTransformer_21431886807859
// MI455X (gfx1250) — matrix operand order repaired and checked by reading, not compiled, not run on hardware
//
#include <hip/hip_runtime.h>
#include <hip/hip_bf16.h>

// ---------------------------------------------------------------------------
// GraphTransformer forward for MI455X (gfx1250), wave32 + WMMA bf16.
// B=64, N=511 nodes (heap tree), D=256, H=8, DK=32, T=5, CLASSES=5.
//
// All 256x256 projections run through a bf16 WMMA GEMM
// (v_wmma_f32_16x16x32_bf16) with 64-row blocking: the A tile is staged into
// LDS via gfx1250 async-to-LDS loads (global_load_async_to_lds_b128 +
// s_wait_asynccnt), and each B fragment is reused across 4 M-tiles to cut L2
// weight traffic 4x. Attention/softmax/layernorm are VALU kernels with wave32
// shuffle reductions. ctx GEMMs are algebraically reduced:
//   kk[:,:,j] = gather_j(h@k1w) + k1b  (children via heap indices),
//   e@{k1w,v1w} precomputed once, state slots are 64-row GEMMs.
// ---------------------------------------------------------------------------

#define BATCH 64
#define NNODE 511
#define DMODEL 256
#define NHEAD 8
#define DKH 32
#define TSTEP 5
#define BN (BATCH * NNODE)          // 32704, multiple of 64

typedef __attribute__((ext_vector_type(16))) __bf16 v16bf;
typedef __attribute__((ext_vector_type(8)))  __bf16 v8bf;
typedef __attribute__((ext_vector_type(8)))  float  v8f;

__device__ __forceinline__ float waveRedSum(float x) {
#pragma unroll
    for (int o = 16; o > 0; o >>= 1) x += __shfl_xor(x, o, 32);
    return x;
}
__device__ __forceinline__ float waveRedMax(float x) {
#pragma unroll
    for (int o = 16; o > 0; o >>= 1) x = fmaxf(x, __shfl_xor(x, o, 32));
    return x;
}

// gfx1250 async copy: LDS[lds_off] = MEM[gaddr], 16 bytes, tracked by ASYNCcnt
__device__ __forceinline__ void async_load_b128(unsigned lds_off, const void* gaddr) {
    asm volatile("global_load_async_to_lds_b128 %0, %1, off"
                 :: "v"(lds_off), "v"(gaddr) : "memory");
}
__device__ __forceinline__ void wait_async0() {
    asm volatile("s_wait_asynccnt 0x0" ::: "memory");
}

// ---------------------------------------------------------------------------
// bf16 GEMM: C[M,256] = A[M,256] @ W[256,256] (+bias). Wt is N-major
// (Wt[n*256+k] = W[k*256+n], bf16). One block = 64 rows; 8 waves, each wave
// computes 4 M-tiles x 2 N-tiles (B fragments reused across the 4 M-tiles).
// Requires M % 64 == 0.
// ---------------------------------------------------------------------------
__global__ __launch_bounds__(256) void gemm256(const __bf16* __restrict__ A,
                                               const __bf16* __restrict__ Wt,
                                               const float* __restrict__ bias,
                                               float* __restrict__ outF,
                                               __bf16* __restrict__ outB) {
    __shared__ __align__(16) __bf16 ldsA[64 * 256];   // 32 KB
    const int t = threadIdx.x;
    const int wave = t >> 5;
    const int lane = t & 31;
    const size_t rowBase = (size_t)blockIdx.x * 64;

    // A tile 64x256 bf16 = 32768 contiguous bytes -> LDS via async engine
    {
        const char* gsrc = (const char*)(A + rowBase * 256);
        const unsigned lbase = (unsigned)(unsigned long long)(void*)ldsA;
#pragma unroll
        for (int i = 0; i < 8; ++i) {
            const unsigned off = (unsigned)t * 16u + (unsigned)i * 4096u;
            async_load_b128(lbase + off, gsrc + off);
        }
        wait_async0();   // own transfers complete; barrier covers other waves
    }
    __syncthreads();

    const int arow  = lane & 15;   // A-matrix row within M-tile for this lane
    const int khalf = lane >> 4;   // lanes 0-15 / 16-31 K-split
    const int ncol  = lane & 15;
    const int nt0 = wave * 2, nt1 = wave * 2 + 1;
    const __bf16* wt0 = Wt + (size_t)(nt0 * 16 + ncol) * 256 + khalf * 16;
    const __bf16* wt1 = Wt + (size_t)(nt1 * 16 + ncol) * 256 + khalf * 16;

    v8f acc[4][2];
#pragma unroll
    for (int mt = 0; mt < 4; ++mt) { acc[mt][0] = (v8f){}; acc[mt][1] = (v8f){}; }

#pragma unroll
    for (int kk = 0; kk < 8; ++kk) {
        // B 32x16: lane holds col (lane&15), 16 contiguous K values (N-major Wt)
        const v16bf b0 = *reinterpret_cast<const v16bf*>(wt0 + kk * 32);
        const v16bf b1 = *reinterpret_cast<const v16bf*>(wt1 + kk * 32);
#pragma unroll
        for (int mt = 0; mt < 4; ++mt) {
            // A in this code's contiguous K order: lane<16 holds K {0..15}, lane>=16 {16..31} (matches B)
            const __bf16* lrow = &ldsA[(mt * 16 + arow) * 256];
            union { v16bf v; v8bf h[2]; } a;
            a.h[0] = *reinterpret_cast<const v8bf*>(lrow + kk * 32 + khalf * 16);
            a.h[1] = *reinterpret_cast<const v8bf*>(lrow + kk * 32 + 8 + khalf * 16);
            acc[mt][0] = __builtin_amdgcn_wmma_f32_16x16x32_bf16(false, a.v, false, b0,
                                                                 (short)0, acc[mt][0], false, false);
            acc[mt][1] = __builtin_amdgcn_wmma_f32_16x16x32_bf16(false, a.v, false, b1,
                                                                 (short)0, acc[mt][1], false, false);
        }
    }

    const int c0 = nt0 * 16 + ncol, c1 = nt1 * 16 + ncol;
    const float bv0 = bias ? bias[c0] : 0.0f;
    const float bv1 = bias ? bias[c1] : 0.0f;
    const int rowOff = khalf * 8;   // C layout: elem r -> row r + 8*(lane>=16)
#pragma unroll
    for (int mt = 0; mt < 4; ++mt) {
#pragma unroll
        for (int r = 0; r < 8; ++r) {
            const size_t row = rowBase + mt * 16 + rowOff + r;
            const float x0 = acc[mt][0][r] + bv0;
            const float x1 = acc[mt][1][r] + bv1;
            if (outF) {
                outF[row * 256 + c0] = x0;
                outF[row * 256 + c1] = x1;
            } else {
                outB[row * 256 + c0] = (__bf16)x0;
                outB[row * 256 + c1] = (__bf16)x1;
            }
        }
    }
}

// ---------------------------------------------------------------------------
// Embedding gather + mask; writes e and initial h (bf16).
// ---------------------------------------------------------------------------
__global__ __launch_bounds__(256) void embed_kernel(const int* __restrict__ wordid,
                                                    const int* __restrict__ mask,
                                                    const float* __restrict__ emb,
                                                    __bf16* __restrict__ ebf,
                                                    __bf16* __restrict__ hbf) {
    const int bn = blockIdx.x;
    const int d = threadIdx.x;
    const int w = wordid[bn];
    const int m = mask[bn];
    const float v = emb[(size_t)(w * m) * 256 + d] * (float)m;
    const __bf16 bv = (__bf16)v;
    ebf[(size_t)bn * 256 + d] = bv;
    hbf[(size_t)bn * 256 + d] = bv;
}

// state = e.mean(axis=1)  -> [B,D] bf16
__global__ __launch_bounds__(256) void init_state(const __bf16* __restrict__ ebf,
                                                  __bf16* __restrict__ statebf) {
    const int t = blockIdx.x * 256 + threadIdx.x;   // B*D threads
    const int b = t >> 8, d = t & 255;
    float s = 0.0f;
    const __bf16* p = ebf + ((size_t)b * NNODE) * 256 + d;
    for (int i = 0; i < NNODE; ++i) s += (float)p[(size_t)i * 256];
    statebf[t] = (__bf16)(s / (float)NNODE);
}

// ---------------------------------------------------------------------------
// Per-node 5-slot attention. One block per node; each wave = one head,
// lanes = DK dims; wave shuffle reductions for q.k dots.
// slots: [cl, cr, h, e, s]; cl/cr are child gathers zeroed at leaves.
// ---------------------------------------------------------------------------
__global__ __launch_bounds__(256) void attn1(const __bf16* __restrict__ Hq,
                                             const __bf16* __restrict__ Hk,
                                             const __bf16* __restrict__ Hv,
                                             const __bf16* __restrict__ Ek,
                                             const __bf16* __restrict__ Ev,
                                             const __bf16* __restrict__ Sk,
                                             const __bf16* __restrict__ Sv,
                                             const float* __restrict__ k1b,
                                             const float* __restrict__ v1b,
                                             __bf16* __restrict__ Obf) {
    const int bn = blockIdx.x;
    const int b = bn / NNODE, i = bn % NNODE;
    const int d = threadIdx.x;
    int li = 2 * i + 1, ri = 2 * i + 2;
    const bool internal = (li < NNODE);
    if (li > NNODE - 1) li = NNODE - 1;
    if (ri > NNODE - 1) ri = NNODE - 1;

    const size_t base = (size_t)bn * 256 + d;
    const size_t bb = (size_t)b * NNODE;
    const float kb = k1b[d], vb = v1b[d];
    const float q = (float)Hq[base];

    float k[5], v[5];
    k[0] = (internal ? (float)Hk[(bb + li) * 256 + d] : 0.0f) + kb;
    k[1] = (internal ? (float)Hk[(bb + ri) * 256 + d] : 0.0f) + kb;
    k[2] = (float)Hk[base] + kb;
    k[3] = (float)Ek[base] + kb;
    k[4] = (float)Sk[(size_t)b * 256 + d];           // bias already in GEMM
    v[0] = (internal ? (float)Hv[(bb + li) * 256 + d] : 0.0f) + vb;
    v[1] = (internal ? (float)Hv[(bb + ri) * 256 + d] : 0.0f) + vb;
    v[2] = (float)Hv[base] + vb;
    v[3] = (float)Ev[base] + vb;
    v[4] = (float)Sv[(size_t)b * 256 + d];

    const float scale = 0.17677669529663688f;        // 1/sqrt(32)
    float sc[5];
#pragma unroll
    for (int j = 0; j < 5; ++j) sc[j] = waveRedSum(q * k[j]) * scale;
    float m = sc[0];
#pragma unroll
    for (int j = 1; j < 5; ++j) m = fmaxf(m, sc[j]);
    float se = 0.0f;
#pragma unroll
    for (int j = 0; j < 5; ++j) { sc[j] = __expf(sc[j] - m); se += sc[j]; }
    const float inv = 1.0f / se;
    float o = 0.0f;
#pragma unroll
    for (int j = 0; j < 5; ++j) o += sc[j] * inv * v[j];
    Obf[base] = (__bf16)o;
}

// relu -> LayerNorm (unbiased std, eps added to std) -> bf16
__global__ __launch_bounds__(256) void relu_ln(const float* __restrict__ in,
                                               const float* __restrict__ g,
                                               const float* __restrict__ bta,
                                               __bf16* __restrict__ out) {
    __shared__ float red[256];
    const int row = blockIdx.x, t = threadIdx.x;
    const float x = fmaxf(in[(size_t)row * 256 + t], 0.0f);
    red[t] = x; __syncthreads();
    for (int s = 128; s > 0; s >>= 1) { if (t < s) red[t] += red[t + s]; __syncthreads(); }
    const float mean = red[0] * (1.0f / 256.0f);
    __syncthreads();
    const float dff = x - mean;
    red[t] = dff * dff; __syncthreads();
    for (int s = 128; s > 0; s >>= 1) { if (t < s) red[t] += red[t + s]; __syncthreads(); }
    const float sd = sqrtf(red[0] * (1.0f / 255.0f));   // ddof=1
    const float y = g[t] * dff / (sd + 1e-6f) + bta[t];
    out[(size_t)row * 256 + t] = (__bf16)y;
}

// ---------------------------------------------------------------------------
// State attention: q=state over keys [h rows ; state]; last slot masked to
// -1e9 (softmaxes to exactly 0 in fp32) so the state key/value rows are
// never materialized. One block per batch, one wave per head.
// ---------------------------------------------------------------------------
__global__ __launch_bounds__(256) void attn2(const __bf16* __restrict__ Sq,
                                             const __bf16* __restrict__ K2h,
                                             const __bf16* __restrict__ V2h,
                                             __bf16* __restrict__ o2in) {
    __shared__ float qs[256];
    __shared__ float sc[NHEAD * 512];
    const int b = blockIdx.x, t = threadIdx.x;
    const int h = t >> 5, lane = t & 31;
    qs[t] = (float)Sq[(size_t)b * 256 + t];
    __syncthreads();

    const float scale = 0.17677669529663688f;
    float scv[16];
#pragma unroll
    for (int kb = 0; kb < 16; ++kb) {
        const int k = kb * 32 + lane;
        float a = -1e9f;                       // slot 511 (state self) masked
        if (k < NNODE) {
            const __bf16* kp = K2h + ((size_t)b * NNODE + k) * 256 + h * 32;
            float acc = 0.0f;
#pragma unroll
            for (int dd = 0; dd < 32; ++dd) acc += qs[h * 32 + dd] * (float)kp[dd];
            a = acc * scale;
        }
        scv[kb] = a;
    }
    float m = scv[0];
#pragma unroll
    for (int kb = 1; kb < 16; ++kb) m = fmaxf(m, scv[kb]);
    m = waveRedMax(m);
    float se = 0.0f;
#pragma unroll
    for (int kb = 0; kb < 16; ++kb) { scv[kb] = __expf(scv[kb] - m); se += scv[kb]; }
    se = waveRedSum(se);
    const float inv = 1.0f / se;
#pragma unroll
    for (int kb = 0; kb < 16; ++kb) sc[h * 512 + kb * 32 + lane] = scv[kb] * inv;
    __syncthreads();

    float acc = 0.0f;
    const __bf16* vp = V2h + (size_t)b * NNODE * 256 + h * 32 + lane;
    for (int k = 0; k < NNODE; ++k) acc += sc[h * 512 + k] * (float)vp[(size_t)k * 256];
    o2in[(size_t)b * 256 + h * 32 + lane] = (__bf16)acc;
}

// Transpose+convert the 8 square weights to N-major bf16 for WMMA B frags.
struct WPtrs { const float* w[8]; };
__global__ __launch_bounds__(256) void prep_weights(WPtrs ps, __bf16* __restrict__ dst) {
    const int mat = blockIdx.x >> 8, n = blockIdx.x & 255, k = threadIdx.x;
    dst[(size_t)mat * 65536 + (size_t)n * 256 + k] = (__bf16)ps.w[mat][(size_t)k * 256 + n];
}

// logits = h @ linw + linb, [BN,256]x[256,5] -> fp32 out. 16 rows per block.
__global__ __launch_bounds__(256) void final_linear(const __bf16* __restrict__ hbf,
                                                    const float* __restrict__ linw,
                                                    const float* __restrict__ linb,
                                                    float* __restrict__ out) {
    __shared__ __align__(16) __bf16 rows[16 * 256];
    const int t = threadIdx.x;
    const size_t r0 = (size_t)blockIdx.x * 16;
    const uint4* src = reinterpret_cast<const uint4*>(hbf + r0 * 256);
    uint4* dst = reinterpret_cast<uint4*>(rows);
    dst[t] = src[t];
    dst[t + 256] = src[t + 256];
    __syncthreads();
    if (t < 80) {
        const int r = t / 5, c = t % 5;
        float acc = linb[c];
        const __bf16* hr = &rows[r * 256];
        for (int d = 0; d < 256; ++d) acc += (float)hr[d] * linw[d * 5 + c];
        out[(r0 + r) * 5 + c] = acc;
    }
}

// ---------------------------------------------------------------------------
extern "C" void kernel_launch(void* const* d_in, const int* in_sizes, int n_in,
                              void* d_out, int out_size, void* d_ws, size_t ws_size,
                              hipStream_t stream) {
    const int*   wordid = (const int*)d_in[0];
    const int*   maskp  = (const int*)d_in[1];
    const float* emb    = (const float*)d_in[2];
    const float* q1w = (const float*)d_in[3],  *q1b = (const float*)d_in[4];
    const float* k1w = (const float*)d_in[5],  *k1b = (const float*)d_in[6];
    const float* v1w = (const float*)d_in[7],  *v1b = (const float*)d_in[8];
    const float* o1w = (const float*)d_in[9],  *o1b = (const float*)d_in[10];
    const float* q2w = (const float*)d_in[11], *q2b = (const float*)d_in[12];
    const float* k2w = (const float*)d_in[13], *k2b = (const float*)d_in[14];
    const float* v2w = (const float*)d_in[15], *v2b = (const float*)d_in[16];
    const float* o2w = (const float*)d_in[17], *o2b = (const float*)d_in[18];
    const float* ln1g = (const float*)d_in[19], *ln1b = (const float*)d_in[20];
    const float* ln2g = (const float*)d_in[21], *ln2b = (const float*)d_in[22];
    const float* linw = (const float*)d_in[23], *linb = (const float*)d_in[24];

    char* ws = (char*)d_ws;
    size_t off = 0;
    auto alloc = [&](size_t bytes) -> char* {
        char* p = ws + off;
        off += (bytes + 255) & ~(size_t)255;
        return p;
    };
    const size_t BnD = (size_t)BN * DMODEL;
    __bf16* ebf     = (__bf16*)alloc(BnD * 2);
    __bf16* hbf     = (__bf16*)alloc(BnD * 2);
    __bf16* Ek      = (__bf16*)alloc(BnD * 2);
    __bf16* Ev      = (__bf16*)alloc(BnD * 2);
    __bf16* Hq      = (__bf16*)alloc(BnD * 2);
    __bf16* Hk      = (__bf16*)alloc(BnD * 2);   // reused as K2h in state phase
    __bf16* Hv      = (__bf16*)alloc(BnD * 2);   // reused as V2h
    __bf16* Obf     = (__bf16*)alloc(BnD * 2);
    float*  tmpF    = (float*) alloc(BnD * 4);
    __bf16* Sk      = (__bf16*)alloc((size_t)BATCH * DMODEL * 2);
    __bf16* Sv      = (__bf16*)alloc((size_t)BATCH * DMODEL * 2);
    __bf16* Sq      = (__bf16*)alloc((size_t)BATCH * DMODEL * 2);
    __bf16* o2in    = (__bf16*)alloc((size_t)BATCH * DMODEL * 2);
    float*  o2tmp   = (float*) alloc((size_t)BATCH * DMODEL * 4);
    __bf16* statebf = (__bf16*)alloc((size_t)BATCH * DMODEL * 2);
    __bf16* wtb     = (__bf16*)alloc((size_t)8 * 65536 * 2);
    // wtb slots: 0=q1w 1=k1w 2=v1w 3=o1w 4=q2w 5=k2w 6=v2w 7=o2w
    __bf16* WQ1 = wtb + 0 * 65536; __bf16* WK1 = wtb + 1 * 65536;
    __bf16* WV1 = wtb + 2 * 65536; __bf16* WO1 = wtb + 3 * 65536;
    __bf16* WQ2 = wtb + 4 * 65536; __bf16* WK2 = wtb + 5 * 65536;
    __bf16* WV2 = wtb + 6 * 65536; __bf16* WO2 = wtb + 7 * 65536;

    const float* FNULL = nullptr;

    WPtrs ps = {{q1w, k1w, v1w, o1w, q2w, k2w, v2w, o2w}};
    prep_weights<<<8 * 256, 256, 0, stream>>>(ps, wtb);
    embed_kernel<<<BN, 256, 0, stream>>>(wordid, maskp, emb, ebf, hbf);
    init_state<<<BATCH, 256, 0, stream>>>(ebf, statebf);
    // loop-invariant e projections (no bias; bias added in attn1)
    gemm256<<<BN / 64, 256, 0, stream>>>(ebf, WK1, FNULL, (float*)nullptr, Ek);
    gemm256<<<BN / 64, 256, 0, stream>>>(ebf, WV1, FNULL, (float*)nullptr, Ev);

    for (int it = 0; it < TSTEP; ++it) {
        gemm256<<<BN / 64, 256, 0, stream>>>(hbf, WQ1, q1b, (float*)nullptr, Hq);
        gemm256<<<BN / 64, 256, 0, stream>>>(hbf, WK1, FNULL, (float*)nullptr, Hk);
        gemm256<<<BN / 64, 256, 0, stream>>>(hbf, WV1, FNULL, (float*)nullptr, Hv);
        gemm256<<<BATCH / 64, 256, 0, stream>>>(statebf, WK1, k1b, (float*)nullptr, Sk);
        gemm256<<<BATCH / 64, 256, 0, stream>>>(statebf, WV1, v1b, (float*)nullptr, Sv);
        attn1<<<BN, 256, 0, stream>>>(Hq, Hk, Hv, Ek, Ev, Sk, Sv, k1b, v1b, Obf);
        gemm256<<<BN / 64, 256, 0, stream>>>(Obf, WO1, o1b, tmpF, (__bf16*)nullptr);
        relu_ln<<<BN, 256, 0, stream>>>(tmpF, ln1g, ln1b, hbf);

        // state update: K2h/V2h over updated h (bias uniform -> in GEMM)
        gemm256<<<BN / 64, 256, 0, stream>>>(hbf, WK2, k2b, (float*)nullptr, Hk);
        gemm256<<<BN / 64, 256, 0, stream>>>(hbf, WV2, v2b, (float*)nullptr, Hv);
        gemm256<<<BATCH / 64, 256, 0, stream>>>(statebf, WQ2, q2b, (float*)nullptr, Sq);
        attn2<<<BATCH, 256, 0, stream>>>(Sq, Hk, Hv, o2in);
        gemm256<<<BATCH / 64, 256, 0, stream>>>(o2in, WO2, o2b, o2tmp, (__bf16*)nullptr);
        relu_ln<<<BATCH, 256, 0, stream>>>(o2tmp, ln2g, ln2b, statebf);
    }

    final_linear<<<BN / 16, 256, 0, stream>>>(hbf, linw, linb, (float*)d_out);
    (void)in_sizes; (void)n_in; (void)out_size; (void)ws_size;
}
